// Attention_33251636806033
// MI455X (gfx1250) — matrix operand order repaired and checked by reading, not compiled, not run on hardware
//
#include <hip/hip_runtime.h>
#include <hip/hip_bf16.h>
#include <stdint.h>

// ---------- types ----------
typedef __attribute__((ext_vector_type(16))) __bf16       v16bf;
typedef __attribute__((ext_vector_type(8)))  float        v8f;
typedef __attribute__((ext_vector_type(4)))  unsigned int u32x4;

union ABFrag { u32x4 q[2]; v16bf v; };      // 32B: one WMMA A/B operand
union U4H    { u32x4 v; unsigned short h[8]; };

__device__ __forceinline__ unsigned short f2bf(float f) {
  union { float f; unsigned u; } v; v.f = f;
  unsigned r = v.u + 0x7FFFu + ((v.u >> 16) & 1u);   // RNE
  return (unsigned short)(r >> 16);
}
__device__ __forceinline__ float bf2f(unsigned short u) {
  union { unsigned u; float f; } v; v.u = ((unsigned)u) << 16;
  return v.f;
}

// Async global->LDS copy of 16B per lane (CDNA5 path, tracked by ASYNCcnt)
__device__ __forceinline__ void async_copy_b128(const unsigned short* gptr,
                                                unsigned short* lptr) {
  unsigned lds = (unsigned)(uintptr_t)lptr;                  // low 32b = LDS offset
  unsigned long long g = (unsigned long long)(uintptr_t)gptr;
  asm volatile("global_load_async_to_lds_b128 %0, %1, off"
               :: "v"(lds), "v"(g) : "memory");
}
__device__ __forceinline__ void wait_async0() {
  asm volatile("s_wait_asynccnt 0x0" ::: "memory");
}

// ---------- 1) f32 -> bf16 convert ----------
__global__ void cvt_f32_bf16(const float* __restrict__ in,
                             unsigned short* __restrict__ out, size_t n) {
  size_t i = (size_t)blockIdx.x * 256 + threadIdx.x;
  if (i < n) out[i] = f2bf(in[i]);
}

// ---------- 2) WMMA GEMM: C[M,N] = A[M,K] @ W[N,K]^T + bias ----------
// Block: 256 thr (8 waves), 128x128 tile, K-step 32, double-buffered LDS
// filled by GLOBAL_LOAD_ASYNC_TO_LDS (overlaps copy with WMMA).
template<bool OUT_F32>
__global__ void __launch_bounds__(256)
gemm_wmma(const unsigned short* __restrict__ A,
          const unsigned short* __restrict__ Bw,
          const float* __restrict__ bias,
          void* __restrict__ out, int M, int N, int K)
{
  __shared__ alignas(16) unsigned short As[2][128 * 40];  // row stride 40 halves = 80B
  __shared__ alignas(16) unsigned short Bs[2][128 * 40];

  const int tid  = threadIdx.x;
  const int lane = tid & 31;
  const int wave = tid >> 5;
  const int wm   = wave >> 1;          // 0..3 (M)
  const int wn   = wave & 1;           // 0..1 (N)
  const int l16  = lane & 15;
  const int half = lane >> 4;          // 0: lanes 0-15, 1: lanes 16-31
  const int bM   = blockIdx.y * 128;
  const int bN   = blockIdx.x * 128;

  // async-stage one 128x32 A tile + one 128x32 B tile into buffer `buf`
  auto stage_async = [&](int buf, int k0) {
    #pragma unroll
    for (int c = tid; c < 512; c += 256) {     // 512 16B chunks each, 2/thread
      int row = c >> 2;
      int col = (c & 3) << 3;                  // halves
      async_copy_b128(A  + (size_t)(bM + row) * K + k0 + col, &As[buf][row * 40 + col]);
      async_copy_b128(Bw + (size_t)(bN + row) * K + k0 + col, &Bs[buf][row * 40 + col]);
    }
  };

  v8f acc[2][4];
  #pragma unroll
  for (int mt = 0; mt < 2; ++mt)
    #pragma unroll
    for (int nt = 0; nt < 4; ++nt)
      #pragma unroll
      for (int r = 0; r < 8; ++r) acc[mt][nt][r] = 0.f;

  stage_async(0, 0);
  wait_async0();
  __syncthreads();

  int cur = 0;
  for (int k0 = 0; k0 < K; k0 += 32) {
    if (k0 + 32 < K) stage_async(cur ^ 1, k0 + 32);   // overlap copy with compute

    // A frags: lane<16 holds K{0..7,16..23}; lane>=16 K{8..15,24..31}
    ABFrag af[2];
    #pragma unroll
    for (int mt = 0; mt < 2; ++mt) {
      const unsigned short* p = &As[cur][(wm * 32 + mt * 16 + l16) * 40 + half * 8];
      af[mt].q[0] = *(const u32x4*)(p);
      af[mt].q[1] = *(const u32x4*)(p + 16);
    }
    // B frags: lane<16 holds K{0..7,16..23} of its N column; lane>=16 K{8..15,24..31}
    ABFrag bfg[4];
    #pragma unroll
    for (int nt = 0; nt < 4; ++nt) {
      const unsigned short* p = &Bs[cur][(wn * 64 + nt * 16 + l16) * 40 + half * 8];
      bfg[nt].q[0] = *(const u32x4*)(p);
      bfg[nt].q[1] = *(const u32x4*)(p + 16);
    }
    #pragma unroll
    for (int mt = 0; mt < 2; ++mt)
      #pragma unroll
      for (int nt = 0; nt < 4; ++nt)
        acc[mt][nt] = __builtin_amdgcn_wmma_f32_16x16x32_bf16(
            false, af[mt].v, false, bfg[nt].v, (short)0, acc[mt][nt], false, false);

    wait_async0();      // next tile landed (only its ops are outstanding)
    __syncthreads();    // all waves done reading cur + writes visible block-wide
    cur ^= 1;
  }

  // epilogue: C layout -> row m = r + half*8, col = lane%16
  #pragma unroll
  for (int mt = 0; mt < 2; ++mt)
    #pragma unroll
    for (int nt = 0; nt < 4; ++nt) {
      int col = bN + wn * 64 + nt * 16 + l16;
      float bv = bias[col];
      #pragma unroll
      for (int r = 0; r < 8; ++r) {
        int row = bM + wm * 32 + mt * 16 + half * 8 + r;
        float v = acc[mt][nt][r] + bv;
        if (OUT_F32) ((float*)out)[(size_t)row * N + col] = v;
        else ((unsigned short*)out)[(size_t)row * N + col] = f2bf(v);
      }
    }
}

// ---------- 3) RoPE + transpose to [B,H,T,hd]; fold 1/sqrt(hd) into Q ----------
__global__ void rope_transpose(const unsigned short* __restrict__ qlin,
                               const unsigned short* __restrict__ klin,
                               const unsigned short* __restrict__ vlin,
                               unsigned short* __restrict__ qr,
                               unsigned short* __restrict__ kr,
                               unsigned short* __restrict__ vr)
{
  size_t idx = (size_t)blockIdx.x * 256 + threadIdx.x;   // over B*T*H*hd = 2^24
  int d = idx & 127;
  int h = (int)(idx >> 7) & 15;
  int t = (int)(idx >> 11) & 2047;
  int b = (int)(idx >> 22);

  float qv = bf2f(qlin[idx]);
  float kv = bf2f(klin[idx]);
  size_t pidx = idx + (d < 64 ? 64 : -64);
  float qp = bf2f(qlin[pidx]);
  float kp = bf2f(klin[pidx]);
  float rq = (d < 64) ? -qp : qp;
  float rk = (d < 64) ? -kp : kp;

  float freq = __expf(-(float)(d & 63) * 0.143911568f);  // ln(10000)/64
  float s, c;
  __sincosf((float)t * freq, &s, &c);

  size_t dst = (((size_t)(b * 16 + h) * 2048) + t) * 128 + d;
  qr[dst] = f2bf((qv * c + rq * s) * 0.088388347f);      // 1/sqrt(128)
  kr[dst] = f2bf(kv * c + rk * s);
  vr[dst] = vlin[idx];
}

// ---------- 4) Flash attention: block = (b,h) x 128 q-rows; wave = 16 q-rows ----------
__global__ void __launch_bounds__(256)
attention_flash(const unsigned short* __restrict__ qr,
                const unsigned short* __restrict__ kr,
                const unsigned short* __restrict__ vr,
                unsigned short* __restrict__ ctx)
{
  __shared__ alignas(16) unsigned short Vt[128 * 40];      // V^T tile: [d][kv], pad stride 40
  __shared__ alignas(16) unsigned short Ps[8 * 16 * 40];   // per-wave P bounce

  const int tid  = threadIdx.x;
  const int lane = tid & 31;
  const int wave = tid >> 5;
  const int l16  = lane & 15;
  const int half = lane >> 4;
  const int bh   = blockIdx.x;                  // 0..63
  const int q0   = blockIdx.y * 128 + wave * 16;
  const size_t base = (size_t)bh * 2048 * 128;  // element base of [T,hd] slab

  // Q fragments (A-layout), one per 32-wide d-step
  ABFrag qf[4];
  {
    const unsigned short* qrow = qr + base + (size_t)(q0 + l16) * 128;
    #pragma unroll
    for (int ks = 0; ks < 4; ++ks) {
      int off = ks * 32 + half * 8;
      qf[ks].q[0] = *(const u32x4*)(qrow + off);
      qf[ks].q[1] = *(const u32x4*)(qrow + off + 16);
    }
  }

  v8f O[8];
  float m[8], l[8];
  #pragma unroll
  for (int nt = 0; nt < 8; ++nt)
    #pragma unroll
    for (int r = 0; r < 8; ++r) O[nt][r] = 0.f;
  #pragma unroll
  for (int r = 0; r < 8; ++r) { m[r] = -3.0e38f; l[r] = 0.f; }

  for (int kv0 = 0; kv0 < 2048; kv0 += 32) {
    // stage V tile transposed: thread -> row tv = tid/8, d-range (tid%8)*16..+15
    {
      int tv = tid >> 3;
      int db = (tid & 7) << 4;
      const unsigned short* vsrc = vr + base + (size_t)(kv0 + tv) * 128 + db;
      U4H a, b;
      a.v = *(const u32x4*)(vsrc);
      b.v = *(const u32x4*)(vsrc + 8);
      #pragma unroll
      for (int j = 0; j < 8; ++j) {
        Vt[(db + j) * 40 + tv]     = a.h[j];
        Vt[(db + 8 + j) * 40 + tv] = b.h[j];
      }
    }
    __syncthreads();

    // S = Q K^T (K rows are contiguous -> direct global B-frags)
    v8f S[2];
    #pragma unroll
    for (int nt = 0; nt < 2; ++nt) {
      #pragma unroll
      for (int r = 0; r < 8; ++r) S[nt][r] = 0.f;
      const unsigned short* krow = kr + base + (size_t)(kv0 + nt * 16 + l16) * 128;
      #pragma unroll
      for (int ks = 0; ks < 4; ++ks) {
        ABFrag kf;
        int off = ks * 32 + half * 8;
        kf.q[0] = *(const u32x4*)(krow + off);
        kf.q[1] = *(const u32x4*)(krow + off + 16);
        S[nt] = __builtin_amdgcn_wmma_f32_16x16x32_bf16(
            false, qf[ks].v, false, kf.v, (short)0, S[nt], false, false);
      }
    }

    // online softmax (rows live in (vgpr r, lane-half); cols across 16-lane group)
    float rmax[8];
    #pragma unroll
    for (int r = 0; r < 8; ++r) rmax[r] = fmaxf(S[0][r], S[1][r]);
    #pragma unroll
    for (int sft = 1; sft < 16; sft <<= 1)
      #pragma unroll
      for (int r = 0; r < 8; ++r) rmax[r] = fmaxf(rmax[r], __shfl_xor(rmax[r], sft, 32));

    float alpha[8], rsum[8];
    #pragma unroll
    for (int r = 0; r < 8; ++r) {
      float mn = fmaxf(m[r], rmax[r]);
      alpha[r] = __expf(m[r] - mn);
      m[r] = mn;
      S[0][r] = __expf(S[0][r] - mn);
      S[1][r] = __expf(S[1][r] - mn);
      rsum[r] = S[0][r] + S[1][r];
    }
    #pragma unroll
    for (int sft = 1; sft < 16; sft <<= 1)
      #pragma unroll
      for (int r = 0; r < 8; ++r) rsum[r] += __shfl_xor(rsum[r], sft, 32);
    #pragma unroll
    for (int r = 0; r < 8; ++r) l[r] = l[r] * alpha[r] + rsum[r];
    #pragma unroll
    for (int nt = 0; nt < 8; ++nt)
      #pragma unroll
      for (int r = 0; r < 8; ++r) O[nt][r] *= alpha[r];

    // bounce P through per-wave LDS: C-layout -> A-layout (wave-synchronous)
    unsigned short* pw = Ps + wave * 16 * 40;
    #pragma unroll
    for (int nt = 0; nt < 2; ++nt)
      #pragma unroll
      for (int r = 0; r < 8; ++r)
        pw[(r + half * 8) * 40 + nt * 16 + l16] = f2bf(S[nt][r]);
    asm volatile("s_wait_dscnt 0x0" ::: "memory");

    ABFrag pf;
    {
      const unsigned short* pr = pw + l16 * 40 + half * 8;
      pf.q[0] = *(const u32x4*)(pr);
      pf.q[1] = *(const u32x4*)(pr + 16);
    }

    // O += P V : V^T rows in LDS give contiguous B-frags
    #pragma unroll
    for (int nt = 0; nt < 8; ++nt) {
      const unsigned short* vtr = Vt + (nt * 16 + l16) * 40 + half * 8;
      ABFrag vf;
      vf.q[0] = *(const u32x4*)(vtr);
      vf.q[1] = *(const u32x4*)(vtr + 16);
      O[nt] = __builtin_amdgcn_wmma_f32_16x16x32_bf16(
          false, pf.v, false, vf.v, (short)0, O[nt], false, false);
    }
    __syncthreads();   // Vt consumed; safe to restage next tile
  }

  // finalize: ctx[(b*T+t), h*128+d] = O/l  (bf16)
  const int b = bh >> 4, h = bh & 15;
  float inv[8];
  #pragma unroll
  for (int r = 0; r < 8; ++r) inv[r] = 1.0f / l[r];
  #pragma unroll
  for (int nt = 0; nt < 8; ++nt)
    #pragma unroll
    for (int r = 0; r < 8; ++r) {
      int t = q0 + half * 8 + r;
      int d = nt * 16 + l16;
      ctx[((size_t)(b * 2048 + t)) * 2048 + h * 128 + d] = f2bf(O[nt][r] * inv[r]);
    }
}

// ---------- launch ----------
extern "C" void kernel_launch(void* const* d_in, const int* in_sizes, int n_in,
                              void* d_out, int out_size, void* d_ws, size_t ws_size,
                              hipStream_t stream)
{
  const float* x    = (const float*)d_in[0];
  const float* wq_w = (const float*)d_in[2];
  const float* wq_b = (const float*)d_in[3];
  const float* wk_w = (const float*)d_in[4];
  const float* wk_b = (const float*)d_in[5];
  const float* wv_w = (const float*)d_in[6];
  const float* wv_b = (const float*)d_in[7];
  const float* wo_w = (const float*)d_in[8];
  const float* wo_b = (const float*)d_in[9];

  const int D = 2048;
  const size_t MT = 8192;                   // B*T
  uint8_t* ws = (uint8_t*)d_ws;
  size_t off = 0;
  auto carve = [&](size_t bytes) -> void* {
    void* p = ws + off;
    off += (bytes + 255) & ~(size_t)255;
    return p;
  };
  unsigned short* xb   = (unsigned short*)carve(MT * D * 2);
  unsigned short* wqb  = (unsigned short*)carve((size_t)D * D * 2);
  unsigned short* wkb  = (unsigned short*)carve((size_t)D * D * 2);
  unsigned short* wvb  = (unsigned short*)carve((size_t)D * D * 2);
  unsigned short* wob  = (unsigned short*)carve((size_t)D * D * 2);
  unsigned short* qlin = (unsigned short*)carve(MT * D * 2);
  unsigned short* klin = (unsigned short*)carve(MT * D * 2);
  unsigned short* vlin = (unsigned short*)carve(MT * D * 2);
  unsigned short* qr   = (unsigned short*)carve(MT * D * 2);
  unsigned short* krp  = (unsigned short*)carve(MT * D * 2);
  unsigned short* vrp  = (unsigned short*)carve(MT * D * 2);
  unsigned short* ctx  = qlin;              // qlin dead after rope -> reuse

  const size_t nX = MT * D, nW = (size_t)D * D;
  cvt_f32_bf16<<<(unsigned)((nX + 255) / 256), 256, 0, stream>>>(x, xb, nX);
  cvt_f32_bf16<<<(unsigned)((nW + 255) / 256), 256, 0, stream>>>(wq_w, wqb, nW);
  cvt_f32_bf16<<<(unsigned)((nW + 255) / 256), 256, 0, stream>>>(wk_w, wkb, nW);
  cvt_f32_bf16<<<(unsigned)((nW + 255) / 256), 256, 0, stream>>>(wv_w, wvb, nW);
  cvt_f32_bf16<<<(unsigned)((nW + 255) / 256), 256, 0, stream>>>(wo_w, wob, nW);

  dim3 gg(D / 128, (unsigned)(MT / 128));
  gemm_wmma<false><<<gg, 256, 0, stream>>>(xb, wqb, wq_b, qlin, (int)MT, D, D);
  gemm_wmma<false><<<gg, 256, 0, stream>>>(xb, wkb, wk_b, klin, (int)MT, D, D);
  gemm_wmma<false><<<gg, 256, 0, stream>>>(xb, wvb, wv_b, vlin, (int)MT, D, D);

  rope_transpose<<<(unsigned)(nX / 256), 256, 0, stream>>>(qlin, klin, vlin, qr, krp, vrp);

  attention_flash<<<dim3(64, 16), 256, 0, stream>>>(qr, krp, vrp, ctx);

  gemm_wmma<true><<<gg, 256, 0, stream>>>(ctx, wob, wo_b, d_out, (int)MT, D, D);
}
